// MultiheadAttentionGate_11647951307117
// MI455X (gfx1250) — matrix operand order repaired and checked by reading, not compiled, not run on hardware
//
#include <hip/hip_runtime.h>

#define HH 8
#define BB 8
#define SS 2048
#define DD 256
#define FF 256
#define BSZ (BB * SS)   // 16384 rows
#define NER 8

typedef __bf16 bf16;
typedef __attribute__((ext_vector_type(16))) __bf16 v16bf;
typedef __attribute__((ext_vector_type(8)))  float  v8f;
typedef __attribute__((ext_vector_type(4)))  unsigned int v4u;
typedef __attribute__((ext_vector_type(4)))  int v4i;
typedef __attribute__((ext_vector_type(8)))  int v8i;

#if defined(__AMDGCN__) && __has_builtin(__builtin_amdgcn_tensor_load_to_lds)
#define USE_TDM 1
#else
#define USE_TDM 0
#endif

__device__ __forceinline__ bf16 f2bf(float f) {
    unsigned u = __float_as_uint(f);
    unsigned r = (u + 0x7fffu + ((u >> 16) & 1u)) >> 16;
    unsigned short s = (unsigned short)r;
    return __builtin_bit_cast(__bf16, s);
}

// A-matrix 16x32 bf16 frag: lane holds row (lane&15); lanes 0-15 K 0..15, lanes 16-31 K 16..31
__device__ __forceinline__ v16bf load_fragA(const bf16* rowp, int lane) {
    const int off = (lane & 16) ? 16 : 0;
    v16bf r;
    reinterpret_cast<uint4*>(&r)[0] = *reinterpret_cast<const uint4*>(rowp + off);
    reinterpret_cast<uint4*>(&r)[1] = *reinterpret_cast<const uint4*>(rowp + off + 8);
    return r;
}
// B-matrix 32x16 bf16 frag: lane holds column (lane&15); lanes 0-15 K 0..15, lanes 16-31 K 16..31
__device__ __forceinline__ v16bf load_fragB(const bf16* colp, int lane) {
    const int off = (lane & 16) ? 16 : 0;
    v16bf r;
    reinterpret_cast<uint4*>(&r)[0] = *reinterpret_cast<const uint4*>(colp + off);
    reinterpret_cast<uint4*>(&r)[1] = *reinterpret_cast<const uint4*>(colp + off + 8);
    return r;
}

__device__ __forceinline__ v8f zero8() {
    v8f z = {0.f, 0.f, 0.f, 0.f, 0.f, 0.f, 0.f, 0.f};
    return z;
}

// ---- Tensor Data Mover: 2D tile global->LDS (D# per CDNA5 ISA) ----
// data_size fixed to 8 bytes (code 3); all dims/strides in 8-byte units.
__device__ __forceinline__ void tdm_load_2d(unsigned lds_byte_off, const void* gptr,
                                            unsigned tensor_d0_units,
                                            unsigned tile_d0_units, unsigned tile_d1,
                                            unsigned long long stride0_units) {
#if USE_TDM
    unsigned long long ga = (unsigned long long)(__SIZE_TYPE__)gptr;
    v4u g0;
    g0[0] = 1u;                                            // count=1, is_restore=0
    g0[1] = lds_byte_off;                                  // lds_addr
    g0[2] = (unsigned)(ga & 0xFFFFFFFFu);                  // global_addr[31:0]
    g0[3] = (unsigned)((ga >> 32) & 0x01FFFFFFu) | (2u << 30); // global_addr[56:32] | type=2
    v8i g1;
    g1[0] = (int)(3u << 16);                               // wg_mask=0, data_size=3 (8B)
    g1[1] = (int)((tensor_d0_units & 0xFFFFu) << 16);      // tensor_dim0[15:0]
    g1[2] = (int)(((tensor_d0_units >> 16) & 0xFFFFu) |    // tensor_dim0[31:16]
                  ((tile_d1 & 0xFFFFu) << 16));            // tensor_dim1[15:0] = tile_d1
    g1[3] = (int)(((tile_d1 >> 16) & 0xFFFFu) |            // tensor_dim1[31:16]
                  ((tile_d0_units & 0xFFFFu) << 16));      // tile_dim0
    g1[4] = (int)(tile_d1 & 0xFFFFu);                      // tile_dim1 (tile_dim2=0)
    g1[5] = (int)(stride0_units & 0xFFFFFFFFull);          // tensor_dim0_stride[31:0]
    g1[6] = (int)((stride0_units >> 32) & 0xFFFFull);      // stride0[47:32], stride1=0
    g1[7] = 0;
    v4i z4 = {0, 0, 0, 0};
#if defined(__clang_major__) && (__clang_major__ >= 23)
    v8i z8 = {0, 0, 0, 0, 0, 0, 0, 0};
    __builtin_amdgcn_tensor_load_to_lds(g0, g1, z4, z4, z8, 0);
#else
    __builtin_amdgcn_tensor_load_to_lds(g0, g1, z4, z4, 0);
#endif
#else
    (void)lds_byte_off; (void)gptr; (void)tensor_d0_units;
    (void)tile_d0_units; (void)tile_d1; (void)stride0_units;
#endif
}

template <int N>
__device__ __forceinline__ void tdm_wait() {
#if USE_TDM
    __builtin_amdgcn_s_wait_tensorcnt(N);   // N is an ICE (template parameter)
#endif
}

// ---------------- kernel 1: convert inputs to bf16 ----------------
__global__ void cvt_inputs(const float* __restrict__ x, const float* __restrict__ l,
                           bf16* __restrict__ xb, bf16* __restrict__ lb) {
    size_t i = (size_t)blockIdx.x * blockDim.x + threadIdx.x;
    if (i < (size_t)BSZ * DD) {
        xb[i] = f2bf(x[i]);
        lb[i] = f2bf(l[i]);
    }
}

// ---------------- kernel 2: transpose+convert weights: wT[which][h][d][f] = W[h][f][d] ----------------
__global__ void cvt_weights(const float* __restrict__ Qw, const float* __restrict__ Kw,
                            const float* __restrict__ Vw, bf16* __restrict__ wT) {
    size_t i = (size_t)blockIdx.x * blockDim.x + threadIdx.x;
    if (i >= (size_t)3 * HH * DD * FF) return;
    int f = (int)(i % FF);
    int d = (int)((i / FF) % DD);
    int h = (int)((i / ((size_t)FF * DD)) % HH);
    int which = (int)(i / ((size_t)FF * DD * HH));
    const float* src = (which == 0) ? Qw : (which == 1) ? Kw : Vw;
    wT[i] = f2bf(src[((size_t)h * FF + f) * DD + d]);
}

// ---------------- kernel 3: QKV projection GEMM (WMMA, W staged in LDS via TDM) ----------------
__global__ void __launch_bounds__(256, 1)
proj_qkv(const bf16* __restrict__ xb, const bf16* __restrict__ lb,
         const bf16* __restrict__ wT,
         const float* __restrict__ Qb, const float* __restrict__ Kb, const float* __restrict__ Vb,
         bf16* __restrict__ qo, bf16* __restrict__ ko, bf16* __restrict__ vt) {
    extern __shared__ __align__(16) unsigned char smem_raw[];
    bf16* Wl = reinterpret_cast<bf16*>(smem_raw);   // 256x256 bf16 = 128KB

    const int which = blockIdx.z / HH;
    const int h     = blockIdx.z % HH;
    const int tid   = threadIdx.x;
    const int lane  = tid & 31;
    const int wave  = tid >> 5;

    const bf16* wsrc = wT + ((size_t)which * HH + h) * DD * FF;
#if USE_TDM
    if (wave == 0) {
        const unsigned lds_base = (unsigned)__builtin_amdgcn_groupstaticsize();
        // 128KB contiguous: 16384 x 1 tile of 8B units
        tdm_load_2d(lds_base, wsrc, 16384u, 16384u, 1u, 16384ull);
        tdm_wait<0>();
    }
#else
    {
        const uint4* s = reinterpret_cast<const uint4*>(wsrc);
        uint4* dsts = reinterpret_cast<uint4*>(smem_raw);
#pragma unroll
        for (int i = 0; i < 32; ++i)
            dsts[tid + i * 256] = s[tid + i * 256];
    }
#endif
    __syncthreads();

    const bf16* A = (which == 0) ? lb : xb;
    const float* bias = (which == 0) ? Qb : (which == 1) ? Kb : Vb;

    const int m0  = blockIdx.x * 128 + wave * 16;
    const int row = m0 + (lane & 15);
    const int boff = (lane & 16) ? 16 : 0;

    v8f acc[16];
#pragma unroll
    for (int t = 0; t < 16; ++t) acc[t] = zero8();

#pragma unroll
    for (int kc = 0; kc < 8; ++kc) {
        v16bf af = load_fragA(A + (size_t)row * FF + kc * 32, lane);
#pragma unroll
        for (int g = 0; g < 4; ++g) {
            // batch 4 B-fragments so ds_loads overlap WMMA issue
            v16bf bfr[4];
#pragma unroll
            for (int u = 0; u < 4; ++u) {
                const bf16* bp = Wl + (size_t)((g * 4 + u) * 16 + (lane & 15)) * FF + kc * 32;
                reinterpret_cast<uint4*>(&bfr[u])[0] = *reinterpret_cast<const uint4*>(bp + boff);
                reinterpret_cast<uint4*>(&bfr[u])[1] = *reinterpret_cast<const uint4*>(bp + boff + 8);
            }
#pragma unroll
            for (int u = 0; u < 4; ++u)
                acc[g * 4 + u] = __builtin_amdgcn_wmma_f32_16x16x32_bf16(
                    false, af, false, bfr[u], (short)0, acc[g * 4 + u], false, false);
        }
    }

    const int half8 = (lane & 16) ? 8 : 0;
#pragma unroll
    for (int nt = 0; nt < 16; ++nt) {
        const int d = nt * 16 + (lane & 15);
        const float bv = bias[h * DD + d];
        if (which < 2) {
            bf16* out = (which == 0) ? qo : ko;
            bf16* p = out + (size_t)h * BSZ * DD;
#pragma unroll
            for (int e = 0; e < 8; ++e) {
                int r = m0 + half8 + e;
                p[(size_t)r * DD + d] = f2bf(acc[nt][e] + bv);
            }
        } else {
            const int b  = m0 / SS;
            const int t0 = (m0 % SS) + half8;
            bf16 tmp[8];
#pragma unroll
            for (int e = 0; e < 8; ++e) tmp[e] = f2bf(acc[nt][e] + bv);
            bf16* p = vt + (((size_t)(h * BB + b)) * DD + d) * SS + t0;
            *reinterpret_cast<uint4*>(p) = *reinterpret_cast<const uint4*>(tmp);
        }
    }
}

// ---------------- kernel 4: flash attention, K/V double-buffered in LDS via TDM ----------------
// grid (SS/128, BB, HH); 8 waves x 16 queries. Per 64-key block: K (64x256 bf16, 32KB)
// and Vt (256x64 bf16, strided tile, 32KB) staged to LDS; 2 buffers = 128KB.
__global__ void __launch_bounds__(256, 1)
flash_attn(const bf16* __restrict__ qb, const bf16* __restrict__ kb,
           const bf16* __restrict__ vt, float* __restrict__ ob) {
    extern __shared__ __align__(16) unsigned char smem_raw[];
    bf16* sm = reinterpret_cast<bf16*>(smem_raw);

    const int h = blockIdx.z, b = blockIdx.y;
    const int tid  = threadIdx.x;
    const int lane = tid & 31;
    const int wave = tid >> 5;
    const int s0   = blockIdx.x * 128 + wave * 16;
    const float scale = 0.0625f;   // 1/sqrt(256)

    const bf16* kbase = kb + ((size_t)h * BSZ + (size_t)b * SS) * DD;
    const bf16* vbase = vt + ((size_t)(h * BB + b)) * DD * SS;

    // q^T fragments (B operand), 16 queries x 256 d
    const bf16* qrow = qb + ((size_t)h * BSZ + (size_t)b * SS + s0 + (lane & 15)) * DD;
    v16bf qf[8];
#pragma unroll
    for (int dc = 0; dc < 8; ++dc) qf[dc] = load_fragB(qrow + dc * 32, lane);

    v8f acc[16];
#pragma unroll
    for (int t = 0; t < 16; ++t) acc[t] = zero8();
    float run_max = -1e30f, run_sum = 0.f;

    const int NB = SS / 64;                 // 32 key blocks
#if USE_TDM
    const unsigned lds_base = (unsigned)__builtin_amdgcn_groupstaticsize();
    auto issueBlk = [&](int blk) {
        const unsigned koff = lds_base + (unsigned)(blk & 1) * 65536u;
        // K block: contiguous 32KB (4096 x 8B)
        tdm_load_2d(koff, kbase + (size_t)blk * 64 * DD, 4096u, 4096u, 1u, 4096ull);
        // Vt block: 256 rows x 128B, row stride = SS*2 bytes = 512 x 8B units
        tdm_load_2d(koff + 32768u, vbase + (size_t)blk * 64, 512u, 16u, 256u, 512ull);
    };
    if (wave == 0) issueBlk(0);
#endif

    for (int blk = 0; blk < NB; ++blk) {
        const int cur = blk & 1;
        const bf16* Kl = sm + (size_t)cur * 32768;  // 64 rows x 256 bf16
        const bf16* Vl = Kl + 16384;                // 256 rows x 64 bf16
#if USE_TDM
        if (wave == 0) {
            if (blk + 1 < NB) { issueBlk(blk + 1); tdm_wait<2>(); }
            else              { tdm_wait<0>(); }
        }
#else
        {   // cooperative stage: 256 threads
            const uint4* ks = reinterpret_cast<const uint4*>(kbase + (size_t)blk * 64 * DD);
            uint4* kd = reinterpret_cast<uint4*>(const_cast<bf16*>(Kl));
#pragma unroll
            for (int i = 0; i < 8; ++i) kd[tid + i * 256] = ks[tid + i * 256];
            const uint4* vs = reinterpret_cast<const uint4*>(vbase + (size_t)tid * SS + blk * 64);
            uint4* vd = reinterpret_cast<uint4*>(const_cast<bf16*>(Vl)) + (size_t)tid * 8;
#pragma unroll
            for (int i = 0; i < 8; ++i) vd[i] = vs[i];
        }
#endif
        __syncthreads();

        // scores^T: 4 tiles of (16 keys x 16 queries) from LDS K
        v8f st[4];
#pragma unroll
        for (int kt = 0; kt < 4; ++kt) {
            v8f s = zero8();
            const bf16* krow = Kl + (size_t)(kt * 16 + (lane & 15)) * DD;
#pragma unroll
            for (int dc = 0; dc < 8; ++dc) {
                v16bf kf = load_fragA(krow + dc * 32, lane);
                s = __builtin_amdgcn_wmma_f32_16x16x32_bf16(
                    false, kf, false, qf[dc], (short)0, s, false, false);
            }
            st[kt] = s;
        }
        // online softmax (per-lane column q = lane&15; halves combined via xor-16)
        float m = run_max;
#pragma unroll
        for (int kt = 0; kt < 4; ++kt)
#pragma unroll
            for (int e = 0; e < 8; ++e) {
                float v = st[kt][e] * scale;
                st[kt][e] = v;
                m = fmaxf(m, v);
            }
        m = fmaxf(m, __shfl_xor(m, 16, 32));
        const float corr = __expf(run_max - m);
        run_max = m;
        run_sum *= corr;
#pragma unroll
        for (int t = 0; t < 16; ++t)
#pragma unroll
            for (int e = 0; e < 8; ++e) acc[t][e] *= corr;
#pragma unroll
        for (int kt = 0; kt < 4; ++kt)
#pragma unroll
            for (int e = 0; e < 8; ++e) {
                float p = __expf(st[kt][e] - m);
                st[kt][e] = p;
                run_sum += p;
            }
        // O^T += Vt-tile (A) x P (B); repack P^T into B layout with one xor-16 exchange
#pragma unroll
        for (int j = 0; j < 2; ++j) {
            v16bf pf;
#pragma unroll
            for (int e = 0; e < 8; ++e) {
                float own_lo = st[2 * j][e];
                float own_hi = st[2 * j + 1][e];
                float send   = (lane < 16) ? own_hi : own_lo;
                float other  = __shfl_xor(send, 16, 32);
                float lo8 = (lane < 16) ? own_lo : other;
                float hi8 = (lane < 16) ? other  : own_hi;
                pf[e]     = f2bf(lo8);
                pf[e + 8] = f2bf(hi8);
            }
#pragma unroll
            for (int dt = 0; dt < 16; ++dt) {
                const bf16* vrow = Vl + (size_t)(dt * 16 + (lane & 15)) * 64 + j * 32;
                v16bf vf = load_fragA(vrow, lane);
                acc[dt] = __builtin_amdgcn_wmma_f32_16x16x32_bf16(
                    false, vf, false, pf, (short)0, acc[dt], false, false);
            }
        }
        __syncthreads();   // all waves done reading this buffer before it is refilled
    }
    run_sum += __shfl_xor(run_sum, 16, 32);
    const float inv = 1.0f / run_sum;

    const int q = lane & 15;
    const int half8 = (lane & 16) ? 8 : 0;
    float* orow = ob + ((size_t)h * BSZ + (size_t)b * SS + s0 + q) * DD;
#pragma unroll
    for (int dt = 0; dt < 16; ++dt) {
        float4 lo, hi;
        lo.x = acc[dt][0] * inv; lo.y = acc[dt][1] * inv;
        lo.z = acc[dt][2] * inv; lo.w = acc[dt][3] * inv;
        hi.x = acc[dt][4] * inv; hi.y = acc[dt][5] * inv;
        hi.z = acc[dt][6] * inv; hi.w = acc[dt][7] * inv;
        *reinterpret_cast<float4*>(orow + dt * 16 + half8)     = lo;
        *reinterpret_cast<float4*>(orow + dt * 16 + half8 + 4) = hi;
    }
}

// ---------------- kernel 5: classifier (N=8, plain FMA) ----------------
__global__ void classifier(const float* __restrict__ ob, const float* __restrict__ Cw,
                           const float* __restrict__ Cb, float* __restrict__ out) {
    int i = blockIdx.x * blockDim.x + threadIdx.x;
    if (i >= BSZ * NER) return;
    int j  = i % NER;
    int bs = i / NER;
    float acc = Cb[j];
#pragma unroll 1
    for (int h = 0; h < HH; ++h) {
        const float* op = ob + ((size_t)h * BSZ + bs) * DD;
        const float* wp = Cw + (size_t)h * DD * NER + j;
#pragma unroll 8
        for (int d = 0; d < DD; ++d)
            acc = fmaf(op[d], wp[(size_t)d * NER], acc);
    }
    out[i] = acc;
}

extern "C" void kernel_launch(void* const* d_in, const int* in_sizes, int n_in,
                              void* d_out, int out_size, void* d_ws, size_t ws_size,
                              hipStream_t stream) {
    (void)in_sizes; (void)n_in; (void)out_size; (void)ws_size;
    const float* x   = (const float*)d_in[0];
    const float* l   = (const float*)d_in[1];
    const float* Qw  = (const float*)d_in[2];
    const float* Qb_ = (const float*)d_in[3];
    const float* Kw  = (const float*)d_in[4];
    const float* Kb_ = (const float*)d_in[5];
    const float* Vw  = (const float*)d_in[6];
    const float* Vb_ = (const float*)d_in[7];
    const float* Cw  = (const float*)d_in[8];
    const float* Cb  = (const float*)d_in[9];
    float* out = (float*)d_out;

    char* ws = (char*)d_ws;
    size_t off = 0;
    auto take = [&](size_t bytes) -> char* {
        off = (off + 255) & ~(size_t)255;
        char* p = ws + off;
        off += bytes;
        return p;
    };
    bf16* xb = (bf16*)take((size_t)BSZ * DD * 2);
    bf16* lb = (bf16*)take((size_t)BSZ * DD * 2);
    bf16* wT = (bf16*)take((size_t)3 * HH * DD * FF * 2);
    bf16* qo = (bf16*)take((size_t)HH * BSZ * DD * 2);
    bf16* ko = (bf16*)take((size_t)HH * BSZ * DD * 2);
    bf16* vt = (bf16*)take((size_t)HH * BSZ * DD * 2);
    float* ob = (float*)take((size_t)HH * BSZ * DD * 4);

    cvt_inputs<<<(BSZ * DD + 255) / 256, 256, 0, stream>>>(x, l, xb, lb);
    cvt_weights<<<(3 * HH * DD * FF + 255) / 256, 256, 0, stream>>>(Qw, Kw, Vw, wT);

    dim3 g1(BSZ / 128, 1, 3 * HH);
    proj_qkv<<<g1, 256, DD * FF * 2, stream>>>(xb, lb, wT, Qb_, Kb_, Vb_, qo, ko, vt);

    dim3 g2(SS / 128, BB, HH);
    flash_attn<<<g2, 256, 131072, stream>>>(qo, ko, vt, ob);

    classifier<<<(BSZ * NER + 255) / 256, 256, 0, stream>>>(ob, Cw, Cb, out);
}
